// AttentionBlock_90787018703078
// MI455X (gfx1250) — matrix operand order repaired and checked by reading, not compiled, not run on hardware
//
#include <hip/hip_runtime.h>

// ---------------------------------------------------------------------------
// AttentionBlock for MI455X (gfx1250): GroupNorm + folded(conv->QKV) WMMA
// GEMMs with ASYNC global->LDS double-buffered staging + f16 LDS-strip
// softmax attention (async-staged K/V panels) + WMMA out-projection +
// residual. Precision: f16 WMMA inputs, f32 accumulation.
// ---------------------------------------------------------------------------

#define B_   16
#define C_   512
#define L_   1024
#define H_   8
#define DK_  64
#define NG_  32
#define CPG_ 16

typedef _Float16 f16;
typedef __attribute__((ext_vector_type(16))) _Float16 v16h;
typedef __attribute__((ext_vector_type(8)))  _Float16 v8h;
typedef __attribute__((ext_vector_type(8)))  float    v8f;

__device__ __forceinline__ v8f wmma_f16(v16h a, v16h b, v8f c) {
  // D = A(16x32,f16) * B(32x16,f16) + C(16x16,f32)
  return __builtin_amdgcn_wmma_f32_16x16x32_f16(false, a, false, b,
                                                (short)0, c, false, false);
}

// CDNA5 async copy: GLOBAL_LOAD_ASYNC_TO_LDS_B128 (ASYNCcnt-tracked, no VGPR
// round-trip). GV addressing: VDST = LDS byte-offset VGPR, VADDR = 64-bit
// global address. Generic pointers to LDS carry the LDS offset in their low
// 32 bits (flat aperture rule, ISA 10.2).
__device__ __forceinline__ void async_b128(void* lds_dst, const void* gsrc) {
  asm volatile("global_load_async_to_lds_b128 %0, %1, off"
               :: "v"((unsigned)(uintptr_t)lds_dst),
                  "v"((unsigned long long)(uintptr_t)gsrc)
               : "memory");
}
__device__ __forceinline__ void wait_async0() {
  asm volatile("s_wait_asynccnt 0" ::: "memory");
}

// Load a 16x32 f16 A-tile from row-major storage (row stride `ld` halves).
// CDNA5 A layout: lane L holds row M=L&15; lanes<16 hold K {0..7,16..23},
// lanes>=16 hold K {8..15,24..31}.
__device__ __forceinline__ v16h load_a_rm(const f16* base, int ld, int lane) {
  const int row = lane & 15;
  const int kb  = (lane >> 4) << 3;
  const f16* p = base + row * ld + kb;
  v8h lo = *(const v8h*)(p);
  v8h hi = *(const v8h*)(p + 16);
  v16h r;
#pragma unroll
  for (int i = 0; i < 8; ++i) { r[i] = lo[i]; r[8 + i] = hi[i]; }
  return r;
}

__device__ __forceinline__ v16h load_b_kn(const f16* base, int ld, int lane) {
  const int col = lane & 15;
  const int kb  = (lane >> 4) << 3;
  v16h r;
#pragma unroll
  for (int i = 0; i < 8; ++i) {
    r[i]     = base[(kb + i) * ld + col];
    r[8 + i] = base[(16 + kb + i) * ld + col];
  }
  return r;
}

// ---------------------------------------------------------------------------
// Shared 64x512 GEMM body: C[64 x 512] = A[64 x 512] * B[512 x 512], with
// async double-buffered LDS staging (K-stage = 64). 8 waves, each owning a
// 16x64 strip per 128-wide column block. Epilogue is a callback.
// ---------------------------------------------------------------------------
template <typename Epi>
__device__ __forceinline__ void gemm64x512(const f16* __restrict__ Xp,
                                           const f16* __restrict__ Wp,
                                           f16* AsBuf /* [2][64*64]  */,
                                           f16* BsBuf /* [2][64*128] */,
                                           Epi epi) {
  const int tid = threadIdx.x, w = tid >> 5, lane = tid & 31;
  // staging assignments (per thread): A: 2 x b128, B: 4 x b128
  const int arow = tid >> 2, akc = (tid & 3) << 4;   // A 64 rows x 64 k
  const int bkk  = tid >> 2, bng = (tid & 3) << 5;   // B 64 k   x 128 n

  for (int n0 = 0; n0 < C_; n0 += 128) {
    v8f acc[4] = {};
    auto stage = [&](int k0, int buf) {
      const f16* xs = &Xp[(size_t)arow * C_ + k0 + akc];
      f16* ad = &AsBuf[buf * (64 * 64) + arow * 64 + akc];
      async_b128(ad, xs);
      async_b128(ad + 8, xs + 8);
      const f16* wsrc = &Wp[(size_t)(k0 + bkk) * C_ + n0 + bng];
      f16* bd = &BsBuf[buf * (64 * 128) + bkk * 128 + bng];
      async_b128(bd,      wsrc);
      async_b128(bd + 8,  wsrc + 8);
      async_b128(bd + 16, wsrc + 16);
      async_b128(bd + 24, wsrc + 24);
      if (k0 + 128 < C_)   // warm L2 two stages ahead
        __builtin_prefetch(&Wp[(size_t)(k0 + 128 + bkk) * C_ + n0 + bng], 0, 1);
    };
    stage(0, 0);
    wait_async0();
    __syncthreads();
    for (int ks = 0; ks < 8; ++ks) {
      const int cur = ks & 1;
      if (ks + 1 < 8) stage((ks + 1) * 64, cur ^ 1);   // overlaps compute below
      const f16* Ab = &AsBuf[cur * (64 * 64) + (w >> 1) * 16 * 64];
      const v16h a0 = load_a_rm(Ab, 64, lane);          // k 0..31 of stage
      const v16h a1 = load_a_rm(Ab + 32, 64, lane);     // k 32..63
      const f16* B0 = &BsBuf[cur * (64 * 128) + (w & 1) * 64];
      const f16* B1 = B0 + 32 * 128;
#pragma unroll
      for (int t = 0; t < 4; ++t) {
        acc[t] = wmma_f16(a0, load_b_kn(B0 + t * 16, 128, lane), acc[t]);
        acc[t] = wmma_f16(a1, load_b_kn(B1 + t * 16, 128, lane), acc[t]);
      }
      wait_async0();      // next stage landed (its copy ran under our WMMAs)
      __syncthreads();
    }
    epi(n0, w, lane, acc);
  }
}

// ---------------------------------------------------------------------------
// 1) Fold 1x1 conv into QKV weights; emit f16 weights in cin-major (B-matrix
//    friendly) layout Wt[s][cin][cout], plus fused f32 biases.
// ---------------------------------------------------------------------------
__global__ __launch_bounds__(256)
void fold_weights_kernel(const float* __restrict__ conv_w, const float* __restrict__ conv_b,
                         const float* __restrict__ wq, const float* __restrict__ bq,
                         const float* __restrict__ wk, const float* __restrict__ bk,
                         const float* __restrict__ wv, const float* __restrict__ bv,
                         const float* __restrict__ wo, const float* __restrict__ bo,
                         f16* __restrict__ Wt, float* __restrict__ bias) {
  const int s = blockIdx.x >> 9;    // 0=q 1=k 2=v 3=o
  const int o = blockIdx.x & 511;
  const float* W = (s == 0) ? wq : (s == 1) ? wk : (s == 2) ? wv : wo;
  const float* Bv = (s == 0) ? bq : (s == 1) ? bk : (s == 2) ? bv : bo;
  const int t = threadIdx.x;
  if (s < 3) {
    for (int c = t; c < C_; c += 256) {
      float acc = 0.f;
#pragma unroll 4
      for (int m = 0; m < C_; ++m) acc += W[o * C_ + m] * conv_w[m * C_ + c];
      Wt[((size_t)s * C_ + c) * C_ + o] = (f16)acc;   // cin-major
    }
    if (t == 0) {
      float acc = Bv[o];
      for (int m = 0; m < C_; ++m) acc += W[o * C_ + m] * conv_b[m];
      bias[s * C_ + o] = acc;
    }
  } else {
    for (int c = t; c < C_; c += 256)
      Wt[((size_t)3 * C_ + c) * C_ + o] = (f16)W[o * C_ + c];
    if (t == 0) bias[3 * C_ + o] = Bv[o];
  }
}

// ---------------------------------------------------------------------------
// 2) GroupNorm over (16ch x 1024px) per (b,g); emit f16 in [B, L, C] layout.
// ---------------------------------------------------------------------------
__global__ __launch_bounds__(256)
void groupnorm_kernel(const float* __restrict__ x, const float* __restrict__ gw,
                      const float* __restrict__ gb, f16* __restrict__ xn) {
  const int b = blockIdx.x >> 5;
  const int g = blockIdx.x & 31;
  const int tid = threadIdx.x;
  const float* xp = x + ((size_t)b * C_ + g * CPG_) * L_;
  float s = 0.f, s2 = 0.f;
  for (int i = tid; i < CPG_ * L_; i += 256) {
    float v = xp[i];
    s += v; s2 += v * v;
  }
  __shared__ float red[2][256];
  red[0][tid] = s; red[1][tid] = s2;
  __syncthreads();
  for (int off = 128; off > 0; off >>= 1) {
    if (tid < off) { red[0][tid] += red[0][tid + off]; red[1][tid] += red[1][tid + off]; }
    __syncthreads();
  }
  const float inv_n = 1.f / (float)(CPG_ * L_);
  const float mean = red[0][0] * inv_n;
  const float var  = red[1][0] * inv_n - mean * mean;
  const float rinv = rsqrtf(var + 1e-5f);
  for (int i = tid; i < CPG_ * L_; i += 256) {
    const int c = g * CPG_ + (i >> 10);
    const int l = i & 1023;
    const float v = (xp[i] - mean) * rinv * gw[c] + gb[c];
    xn[((size_t)b * L_ + l) * C_ + c] = (f16)v;   // transpose to [B,L,C]
  }
}

// ---------------------------------------------------------------------------
// 3) QKV projection GEMM. K output is stored transposed per head: Kt[b,h,d,l].
// ---------------------------------------------------------------------------
__global__ __launch_bounds__(256)
void qkv_gemm_kernel(const f16* __restrict__ XN, const f16* __restrict__ Wt,
                     const float* __restrict__ bias,
                     f16* __restrict__ Qb, f16* __restrict__ Ktb, f16* __restrict__ Vb) {
  const int s  = blockIdx.y;             // 0=q 1=k 2=v
  const int b  = blockIdx.x >> 4;
  const int m0 = (blockIdx.x & 15) * 64;
  __shared__ f16 AsBuf[2 * 64 * 64];
  __shared__ f16 BsBuf[2 * 64 * 128];
  const f16* Xp = XN + ((size_t)b * L_ + m0) * C_;
  const f16* Wp = Wt + (size_t)s * C_ * C_;   // cin-major: [k][n]

  gemm64x512(Xp, Wp, AsBuf, BsBuf,
    [&](int n0, int w, int lane, v8f (&acc)[4]) {
      const int colbase = n0 + (w & 1) * 64;
      const int rowbase = m0 + (w >> 1) * 16;
#pragma unroll
      for (int t = 0; t < 4; ++t)
#pragma unroll
        for (int rr = 0; rr < 8; ++rr) {
          const int row = rowbase + ((lane >> 4) << 3) + rr;
          const int col = colbase + t * 16 + (lane & 15);
          const float val = acc[t][rr] + bias[s * C_ + col];
          if (s == 1) {            // K: store transposed per head
            const int hh = col >> 6, dd = col & 63;
            Ktb[(((size_t)b * H_ + hh) * DK_ + dd) * L_ + row] = (f16)val;
          } else {
            f16* O = (s == 0) ? Qb : Vb;
            O[((size_t)b * L_ + row) * C_ + col] = (f16)val;
          }
        }
    });
}

// ---------------------------------------------------------------------------
// 4) Attention: one block per (b, h, 64-row q-tile). 64x1024 f16 score strip
//    in LDS (128 KB). Two-pass softmax writes back unnormalized exp(s-max);
//    the per-row 1/sum folds into the output accumulator. K and V panels are
//    async-staged through a shared 32 KB double buffer; the first V copy is
//    issued before softmax so it overlaps the whole reduction phase.
// ---------------------------------------------------------------------------
__global__ __launch_bounds__(256)
void attention_kernel(const f16* __restrict__ Q, const f16* __restrict__ Kt,
                      const f16* __restrict__ V, f16* __restrict__ AO) {
  extern __shared__ char smem[];
  f16*   Sh      = (f16*)smem;                           // [64][1024] f16
  f16*   Ts      = (f16*)(smem + 64 * 1024 * 2);         // [2][64*128] staging
  float* red     = (float*)(smem + 64 * 1024 * 2 + 2 * 8192 * 2);  // [64][4]
  float* rowstat = red + 64 * 4;                         // [64] 1/rowsum

  const int bh = blockIdx.x >> 4;
  const int b = bh >> 3, h = bh & 7;
  const int m0 = (blockIdx.x & 15) * 64;
  const int tid = threadIdx.x, w = tid >> 5, lane = tid & 31;
  const int r = w >> 1, half = w & 1;           // rowtile 0..3, col parity
  const float scale = 0.125f;                   // 1/sqrt(64)

  const f16* Qp  = Q  + ((size_t)b * L_ + m0 + r * 16) * C_ + h * DK_;
  const f16* Ktp = Kt + (size_t)bh * DK_ * L_;  // [d][l]
  const f16* Vp  = V  + (size_t)b * L_ * C_ + h * DK_;

  // Q A-tiles for this wave's 16 rows, kept in VGPRs for the whole kernel.
  const v16h qa0 = load_a_rm(Qp, C_, lane);
  const v16h qa1 = load_a_rm(Qp + 32, C_, lane);

  // async staging maps (each thread: 4 x b128 = 64 B of a contiguous row)
  auto stageK = [&](int lb, int buf) {          // Kt panel: [64 d][128 l]
    const f16* src = Ktp + (size_t)(tid >> 1) * L_ + lb + ((tid & 1) << 5);
    f16* dst = Ts + buf * 8192 + (tid >> 1) * 128 + ((tid & 1) << 5);
    async_b128(dst, src);           async_b128(dst + 8, src + 8);
    async_b128(dst + 16, src + 16); async_b128(dst + 24, src + 24);
  };
  auto stageV = [&](int keyb, int buf) {        // V panel: [128 key][64 d]
    const f16* src = Vp + (size_t)(keyb + (tid >> 1)) * C_ + ((tid & 1) << 5);
    f16* dst = Ts + buf * 8192 + (tid >> 1) * 64 + ((tid & 1) << 5);
    async_b128(dst, src);           async_b128(dst + 8, src + 8);
    async_b128(dst + 16, src + 16); async_b128(dst + 24, src + 24);
  };

  // ---- phase 1: S = scale * Q K^T -> Sh (f16 logits) ----
  stageK(0, 0);
  wait_async0(); __syncthreads();
  for (int ls = 0; ls < 8; ++ls) {              // 128-key panels
    const int cur = ls & 1;
    if (ls + 1 < 8) stageK((ls + 1) * 128, cur ^ 1);
    const f16* Kbuf = Ts + cur * 8192;
    for (int jj = half; jj < 8; jj += 2) {      // 16-key tiles within panel
      v8f acc = {};
      const v16h b0 = load_b_kn(Kbuf + jj * 16, 128, lane);
      const v16h b1 = load_b_kn(Kbuf + 32 * 128 + jj * 16, 128, lane);
      acc = wmma_f16(qa0, b0, acc);
      acc = wmma_f16(qa1, b1, acc);
      const int col = (ls * 8 + jj) * 16 + (lane & 15);
#pragma unroll
      for (int rr = 0; rr < 8; ++rr) {
        const int row = r * 16 + ((lane >> 4) << 3) + rr;
        Sh[row * 1024 + col] = (f16)(acc[rr] * scale);
      }
    }
    wait_async0(); __syncthreads();
  }

  stageV(0, 0);   // V panel 0 copy runs underneath the softmax phase

  // ---- phase 2: row stats; Sh <- exp(s - rowmax), unnormalized ----
  const int srow = tid >> 2, q4 = tid & 3;
  f16* Shr = Sh + srow * 1024 + q4 * 256;
  float mx = -1e30f;
  for (int i = 0; i < 256; i += 8) {
    const v8h v = *(const v8h*)(Shr + i);
#pragma unroll
    for (int k = 0; k < 8; ++k) mx = fmaxf(mx, (float)v[k]);
  }
  red[srow * 4 + q4] = mx;
  __syncthreads();
  const float rmax = fmaxf(fmaxf(red[srow * 4 + 0], red[srow * 4 + 1]),
                           fmaxf(red[srow * 4 + 2], red[srow * 4 + 3]));
  __syncthreads();
  float sum = 0.f;
  for (int i = 0; i < 256; i += 8) {
    const v8h v = *(const v8h*)(Shr + i);
    v8h e;
#pragma unroll
    for (int k = 0; k < 8; ++k) {
      const float ef = __expf((float)v[k] - rmax);
      sum += ef; e[k] = (f16)ef;
    }
    *(v8h*)(Shr + i) = e;
  }
  red[srow * 4 + q4] = sum;
  __syncthreads();
  if (q4 == 0)
    rowstat[srow] = 1.0f / (red[srow * 4 + 0] + red[srow * 4 + 1] +
                            red[srow * 4 + 2] + red[srow * 4 + 3]);

  // ---- phase 3: O = (1/rowsum) * E @ V ----
  const int dt0 = (w & 1) * 2;                  // d tiles {dt0, dt0+1}
  v8f oa = {}, ob = {};
  wait_async0(); __syncthreads();               // V panel 0 ready; stats visible
  for (int vs = 0; vs < 8; ++vs) {              // 128-key panels
    const int cur = vs & 1;
    if (vs + 1 < 8) stageV((vs + 1) * 128, cur ^ 1);
    const f16* Vbuf = Ts + cur * 8192;
#pragma unroll
    for (int ki = 0; ki < 4; ++ki) {            // 32-key WMMA K-steps
      const v16h pa =
          load_a_rm(Sh + (r * 16) * 1024 + vs * 128 + ki * 32, 1024, lane);
      const f16* vb = Vbuf + (ki * 32) * 64 + dt0 * 16;
      oa = wmma_f16(pa, load_b_kn(vb, 64, lane), oa);
      ob = wmma_f16(pa, load_b_kn(vb + 16, 64, lane), ob);
    }
    wait_async0(); __syncthreads();
  }
#pragma unroll
  for (int rr = 0; rr < 8; ++rr) {
    const int lrow = r * 16 + ((lane >> 4) << 3) + rr;
    const float rin = rowstat[lrow];
    const int row = m0 + lrow;
    const int c0  = h * DK_ + dt0 * 16 + (lane & 15);
    AO[((size_t)b * L_ + row) * C_ + c0]      = (f16)(oa[rr] * rin);
    AO[((size_t)b * L_ + row) * C_ + c0 + 16] = (f16)(ob[rr] * rin);
  }
}

// ---------------------------------------------------------------------------
// 5) Output projection + bias + residual; writes f32 [B,C,H,W].
// ---------------------------------------------------------------------------
__global__ __launch_bounds__(256)
void outproj_kernel(const f16* __restrict__ AO, const f16* __restrict__ Wt3,
                    const float* __restrict__ bias3, const float* __restrict__ x,
                    float* __restrict__ out) {
  const int b  = blockIdx.x >> 4;
  const int m0 = (blockIdx.x & 15) * 64;
  __shared__ f16 AsBuf[2 * 64 * 64];
  __shared__ f16 BsBuf[2 * 64 * 128];
  const f16* Ap = AO + ((size_t)b * L_ + m0) * C_;

  gemm64x512(Ap, Wt3, AsBuf, BsBuf,
    [&](int n0, int w, int lane, v8f (&acc)[4]) {
      const int colbase = n0 + (w & 1) * 64;
      const int rowbase = m0 + (w >> 1) * 16;
#pragma unroll
      for (int t = 0; t < 4; ++t)
#pragma unroll
        for (int rr = 0; rr < 8; ++rr) {
          const int row = rowbase + ((lane >> 4) << 3) + rr;
          const int col = colbase + t * 16 + (lane & 15);
          const size_t xi = ((size_t)b * C_ + col) * L_ + row;   // [B,C,L]
          out[xi] = acc[t][rr] + bias3[col] + x[xi];
        }
    });
}

// ---------------------------------------------------------------------------
extern "C" void kernel_launch(void* const* d_in, const int* in_sizes, int n_in,
                              void* d_out, int out_size, void* d_ws, size_t ws_size,
                              hipStream_t stream) {
  (void)in_sizes; (void)n_in; (void)out_size; (void)ws_size;
  const float* x      = (const float*)d_in[0];
  const float* gn_w   = (const float*)d_in[1];
  const float* gn_b   = (const float*)d_in[2];
  const float* conv_w = (const float*)d_in[3];
  const float* conv_b = (const float*)d_in[4];
  const float* wq = (const float*)d_in[5];  const float* bq = (const float*)d_in[6];
  const float* wk = (const float*)d_in[7];  const float* bk = (const float*)d_in[8];
  const float* wv = (const float*)d_in[9];  const float* bv = (const float*)d_in[10];
  const float* wo = (const float*)d_in[11]; const float* bo = (const float*)d_in[12];
  float* out = (float*)d_out;

  const size_t ACT = (size_t)B_ * L_ * C_ * sizeof(f16);   // 16 MB each
  char* p = (char*)d_ws;
  f16*   Wt   = (f16*)p;   p += (size_t)4 * C_ * C_ * sizeof(f16);
  float* bias = (float*)p; p += (size_t)4 * C_ * sizeof(float);
  f16*   XN   = (f16*)p;   p += ACT;
  f16*   Qb   = (f16*)p;   p += ACT;
  f16*   Ktb  = (f16*)p;   p += ACT;
  f16*   Vb   = (f16*)p;   p += ACT;
  f16*   AOb  = (f16*)p;   p += ACT;

  fold_weights_kernel<<<dim3(4 * C_), 256, 0, stream>>>(
      conv_w, conv_b, wq, bq, wk, bk, wv, bv, wo, bo, Wt, bias);
  groupnorm_kernel<<<dim3(B_ * NG_), 256, 0, stream>>>(x, gn_w, gn_b, XN);
  qkv_gemm_kernel<<<dim3(B_ * 16, 3), 256, 0, stream>>>(XN, Wt, bias, Qb, Ktb, Vb);
  // 128 KB strip + 32 KB staging + stats
  const size_t att_lds = (size_t)(64 * 1024 * 2 + 2 * 8192 * 2 +
                                  (64 * 4 + 64) * 4);
  attention_kernel<<<dim3(B_ * H_ * 16), 256, att_lds, stream>>>(Qb, Ktb, Vb, AOb);
  outproj_kernel<<<dim3(B_ * 16), 256, 0, stream>>>(
      AOb, Wt + (size_t)3 * C_ * C_, bias + 3 * C_, x, out);
}
